// MultiSourceAttentionBlock_9612136809030
// MI455X (gfx1250) — compile-verified, matrix operand order checked by reading, not run on hardware
//
#include <hip/hip_runtime.h>
#include <math.h>

// ---------------------------------------------------------------------------
// MI455X (gfx1250) implementation of the multi-source attention block.
// - All GEMM-shaped work (13 projections + QK^T + P@V) on v_wmma_f32_16x16x32_bf16
//   (wave32 WMMA, fp32 accumulate, bf16 operands).
// - Attention P@V stages V tiles LDS-side with GLOBAL_LOAD_ASYNC_TO_LDS_B128
//   (ASYNCcnt / s_wait_asynccnt, CDNA5 async-memory path) and reads softmax
//   probabilities as bf16 straight from LDS (ds_load_b128 fragments).
// - Elementwise/normalization work (adaLN, LN, RMS+RoPE, softmax, residuals)
//   is fp32 VALU and bandwidth-trivial at 23.3 TB/s.
// ---------------------------------------------------------------------------

#define B_   4
#define T_   512
#define H_   16
#define HD_  64
#define C_   1024
#define DFF_ 4096
#define TR_  256
#define TCOLS 1792        // 512+512+512+256
#define SROW  1800        // padded LDS score row stride (floats)

typedef __attribute__((ext_vector_type(16))) __bf16 v16bf;
typedef __attribute__((ext_vector_type(8)))  float  v8f;

union __align__(16) Frag {
  v16bf v;
  unsigned short u[16];
  uint4 q2[2];
};

__device__ __forceinline__ unsigned short f2bfu(float f) {
  union { float f; unsigned u; } x; x.f = f;
  unsigned r = x.u + 0x7fffu + ((x.u >> 16) & 1u);   // round-to-nearest-even
  return (unsigned short)(r >> 16);
}

// ---------------------------------------------------------------------------
// adaLN: mod[b, 0:6144] = silu(cond[b]) @ W + bias
// ---------------------------------------------------------------------------
__global__ void adaln_kernel(const float* __restrict__ cond,
                             const float* __restrict__ w,
                             const float* __restrict__ b,
                             float* __restrict__ mod) {
  __shared__ float sc[C_];
  int bb = blockIdx.x / 24;                 // 6144/256 = 24 blocks per batch
  int n  = (blockIdx.x % 24) * 256 + threadIdx.x;
  for (int i = threadIdx.x; i < C_; i += 256) {
    float c = cond[bb * C_ + i];
    sc[i] = c / (1.f + expf(-c));
  }
  __syncthreads();
  float s = 0.f;
  for (int k = 0; k < C_; ++k) s += sc[k] * w[(size_t)k * (6 * C_) + n];
  mod[bb * (6 * C_) + n] = s + b[n];
}

// ---------------------------------------------------------------------------
// LayerNorm + adaLN modulate: out = LN(in)*(1+scale)+shift
// ---------------------------------------------------------------------------
__global__ void ln_mod_kernel(const float* __restrict__ in,
                              const float* __restrict__ w,
                              const float* __restrict__ b,
                              const float* __restrict__ mod,
                              int scale_idx, int shift_idx,
                              float* __restrict__ out) {
  int row = blockIdx.x;                     // B*T rows
  int bb  = row >> 9;                       // / T_
  const float* xr = in + (size_t)row * C_;
  float s0 = 0.f, s1 = 0.f;
  for (int c = threadIdx.x; c < C_; c += 256) { float v = xr[c]; s0 += v; s1 += v * v; }
  __shared__ float r0[256], r1[256];
  r0[threadIdx.x] = s0; r1[threadIdx.x] = s1;
  __syncthreads();
  for (int st = 128; st; st >>= 1) {
    if (threadIdx.x < st) { r0[threadIdx.x] += r0[threadIdx.x + st];
                            r1[threadIdx.x] += r1[threadIdx.x + st]; }
    __syncthreads();
  }
  float mu  = r0[0] * (1.f / C_);
  float var = r1[0] * (1.f / C_) - mu * mu;
  float inv = rsqrtf(var + 1e-5f);
  const float* scl = mod + bb * (6 * C_) + scale_idx * C_;
  const float* shf = mod + bb * (6 * C_) + shift_idx * C_;
  float* orow = out + (size_t)row * C_;
  for (int c = threadIdx.x; c < C_; c += 256) {
    float v = (xr[c] - mu) * inv * w[c] + b[c];
    orow[c] = v * (1.f + scl[c]) + shf[c];
  }
}

// ---------------------------------------------------------------------------
// Tiled WMMA GEMM: C[M,N] = epi(A[M,K] @ W[K,N] + bias) (+ add)
// 64x64 tile / block, 4 waves, each wave = 16(M) x 64(N), BK = 32.
// EPI: 0 = none, 1 = tanh, 2 = exact GELU
// ---------------------------------------------------------------------------
template <int EPI>
__global__ __launch_bounds__(128) void gemm_wmma(
    const float* __restrict__ A, const float* __restrict__ W,
    const float* __restrict__ bias, const float* __restrict__ add,
    float* __restrict__ C, int M, int N, int K) {
  __shared__ __align__(16) unsigned short As[64 * 32];   // [m][k]
  __shared__ __align__(16) unsigned short Bs[64 * 32];   // [n][k] (transposed)
  const int tid  = threadIdx.x;
  const int lane = tid & 31, wave = tid >> 5;
  const int lr   = lane & 15, lh   = lane >> 4;
  const int m0 = blockIdx.y * 64, n0 = blockIdx.x * 64;

  v8f acc[4] = {};
  for (int k0 = 0; k0 < K; k0 += 32) {
    for (int i = tid; i < 2048; i += 128) {
      int r = i >> 5, c = i & 31;
      As[i] = f2bfu(A[(size_t)(m0 + r) * K + (k0 + c)]);
      Bs[i] = f2bfu(W[(size_t)(k0 + c) * N + (n0 + r)]);
    }
    __syncthreads();
    if (k0 + 32 < K) {
      __builtin_prefetch(&A[(size_t)(m0 + (tid >> 1)) * K + k0 + 32], 0, 1);
      __builtin_prefetch(&W[(size_t)(k0 + 32 + (tid & 31)) * N + n0], 0, 1);
    }
    Frag Af;
    const unsigned short* arow = &As[(wave * 16 + lr) * 32];
    Af.q2[0] = *(const uint4*)(arow + 8 * lh);
    Af.q2[1] = *(const uint4*)(arow + 16 + 8 * lh);
#pragma unroll
    for (int j = 0; j < 4; ++j) {
      Frag Bf;
      const unsigned short* brow = &Bs[(j * 16 + lr) * 32];
      Bf.q2[0] = *(const uint4*)(brow + 8 * lh);
      Bf.q2[1] = *(const uint4*)(brow + 16 + 8 * lh);
      acc[j] = __builtin_amdgcn_wmma_f32_16x16x32_bf16(
          false, Af.v, false, Bf.v, (short)0, acc[j], false, false);
    }
    __syncthreads();
  }
#pragma unroll
  for (int j = 0; j < 4; ++j) {
    int col = n0 + j * 16 + lr;
    float bv = bias[col];
#pragma unroll
    for (int r = 0; r < 8; ++r) {
      size_t row = (size_t)(m0 + wave * 16 + r + 8 * lh);
      float v = acc[j][r] + bv;
      if (EPI == 1)      v = tanhf(v);
      else if (EPI == 2) v = 0.5f * v * (1.f + erff(v * 0.70710678f));
      if (add) v += add[row * N + col];
      C[row * N + col] = v;
    }
  }
}

// ---------------------------------------------------------------------------
// Q/K/V prep: (B,L,1024) fp32 -> bf16 [B][H][L][64]; optionally per-head
// RMSNorm (weight) + RoPE. One wave per (b,l,h); lane handles d and d+32.
// ---------------------------------------------------------------------------
template <int DO_RR>
__global__ void qk_prep(const float* __restrict__ in,
                        const float* __restrict__ rmsw,
                        unsigned short* __restrict__ out, int L) {
  int wid  = blockIdx.x * (blockDim.x >> 5) + (threadIdx.x >> 5);
  int lane = threadIdx.x & 31;
  int h = wid % H_;
  int l = (wid / H_) % L;
  int b = wid / (H_ * L);
  const float* row = in + ((size_t)(b * L + l)) * C_ + h * HD_;
  float x0 = row[lane], x1 = row[lane + 32];
  float o0, o1;
  if (DO_RR) {
    float ss = x0 * x0 + x1 * x1;
    for (int m = 16; m; m >>= 1) ss += __shfl_xor(ss, m, 32);
    float r  = rsqrtf(ss * (1.f / HD_) + 1e-6f);
    float y0 = x0 * r * rmsw[lane], y1 = x1 * r * rmsw[lane + 32];
    float fr = (float)l * expf(-(float)(2 * lane) * (1.f / HD_) * 9.2103403720f);
    float cs = __cosf(fr), sn = __sinf(fr);
    o0 = y0 * cs - y1 * sn;       // d < 32 pairs with d+32
    o1 = y1 * cs + y0 * sn;
  } else {
    o0 = x0; o1 = x1;
  }
  unsigned short* orow = out + ((size_t)((b * H_ + h) * L + l)) * HD_;
  orow[lane] = f2bfu(o0);
  orow[lane + 32] = f2bfu(o1);
}

// ---------------------------------------------------------------------------
// Fused 4-source attention. Grid (T/16, H, B), 128 threads (4 waves).
// Dynamic LDS layout:
//   S  : fp32 scores   [16][SROW]                 (115200 B)
//   Pb : bf16 probs    [16][TCOLS]                ( 57344 B)
//   Vst: bf16 V stage  [4 waves][32 s][16 d]      (  4096 B)
// ---------------------------------------------------------------------------
__global__ __launch_bounds__(128) void attn_kernel(
    const unsigned short* __restrict__ qS, const unsigned short* __restrict__ qB,
    const unsigned short* __restrict__ qV, const unsigned short* __restrict__ qR,
    const unsigned short* __restrict__ kS, const unsigned short* __restrict__ kB,
    const unsigned short* __restrict__ kV, const unsigned short* __restrict__ kR,
    const unsigned short* __restrict__ vS, const unsigned short* __restrict__ vB,
    const unsigned short* __restrict__ vV, const unsigned short* __restrict__ vR,
    const float* gating, const float* tSelf, const float* tBev,
    const float* tVl, const float* tRe,
    const float* bSelf, const float* bBev, const float* bVl, const float* bRe,
    float* __restrict__ out) {
  extern __shared__ float smem[];
  float* S = smem;                                            // [16][SROW]
  unsigned short* Pb  = (unsigned short*)(S + 16 * SROW);     // [16][TCOLS]
  unsigned short* Vst = Pb + 16 * TCOLS;                      // [4][32][16]
  const int q0 = blockIdx.x * 16, h = blockIdx.y, b = blockIdx.z;
  const int tid = threadIdx.x, lane = tid & 31, wave = tid >> 5;
  const int lr = lane & 15, lh = lane >> 4;

  const float g  = tanhf(gating[0]);
  const float sc = 0.125f;                  // 1/sqrt(64)
  const float aa[4] = { sc / (1.f + log1pf(expf(tSelf[0]))),
                        sc / (1.f + log1pf(expf(tBev[0]))),
                        sc / (1.f + log1pf(expf(tVl[0]))),
                        g * sc / (1.f + log1pf(expf(tRe[0]))) };
  const float cc[4] = { bSelf[0], bBev[0], bVl[0], g * bRe[0] };
  const unsigned short* Qt[4] = { qS, qB, qV, qR };
  const unsigned short* Kt[4] = { kS, kB, kV, kR };
  const unsigned short* Vt[4] = { vS, vB, vV, vR };
  const int Lsrc[4] = { 512, 512, 512, 256 };
  const int coff[4] = { 0, 512, 1024, 1536 };

  // --- QK^T: 112 key tiles of 16, wave-strided -----------------------------
  for (int tt = wave; tt < 112; tt += 4) {
    int src = (tt < 32) ? 0 : (tt < 64) ? 1 : (tt < 96) ? 2 : 3;
    int s0  = (tt - ((src == 0) ? 0 : (src == 1) ? 32 : (src == 2) ? 64 : 96)) * 16;
    const unsigned short* Qp = Qt[src] + (((size_t)(b * H_ + h)) * T_ + q0 + lr) * HD_;
    const unsigned short* Kp = Kt[src] + (((size_t)(b * H_ + h)) * Lsrc[src] + s0 + lr) * HD_;
    v8f acc = {};
#pragma unroll
    for (int dc = 0; dc < HD_; dc += 32) {
      Frag Af, Bf;
      Af.q2[0] = *(const uint4*)(Qp + dc + 8 * lh);       // A: 16x32 of Q
      Af.q2[1] = *(const uint4*)(Qp + dc + 16 + 8 * lh);
      Bf.q2[0] = *(const uint4*)(Kp + dc + 8 * lh);       // B: K^T (d-major/lane)
      Bf.q2[1] = *(const uint4*)(Kp + dc + 16 + 8 * lh);
      acc = __builtin_amdgcn_wmma_f32_16x16x32_bf16(
          false, Af.v, false, Bf.v, (short)0, acc, false, false);
    }
    int colb = coff[src] + s0 + lr;
#pragma unroll
    for (int r = 0; r < 8; ++r)
      S[(r + 8 * lh) * SROW + colb] = acc[r] * aa[src] + cc[src];
  }
  __syncthreads();

  // --- softmax over 1792 columns, 8 lanes per row; emit bf16 probs ---------
  {
    int row = tid >> 3, j = tid & 7;
    float mx = -1e30f;
    for (int c = j; c < TCOLS; c += 8) mx = fmaxf(mx, S[row * SROW + c]);
    for (int m = 1; m < 8; m <<= 1) mx = fmaxf(mx, __shfl_xor(mx, m, 32));
    float sum = 0.f;
    for (int c = j; c < TCOLS; c += 8) {
      float e = __expf(S[row * SROW + c] - mx);
      S[row * SROW + c] = e; sum += e;
    }
    for (int m = 1; m < 8; m <<= 1) sum += __shfl_xor(sum, m, 32);
    float inv = 1.f / sum;
    for (int c = j; c < TCOLS; c += 8)
      Pb[row * TCOLS + c] = f2bfu(S[row * SROW + c] * inv);
  }
  __syncthreads();

  // --- P @ V_concat: wave owns 16 output dims, 56 chunks of 32 keys --------
  // Per chunk: each lane async-copies one 32-byte V row slice into the wave's
  // private LDS stage (CDNA5 GLOBAL_LOAD_ASYNC_TO_LDS_B128, ASYNCcnt-tracked),
  // then fragments are built from LDS.
  unsigned short* Vw = Vst + wave * (32 * 16);               // [s][d0..15]
  const unsigned vw_lds = (unsigned)(size_t)(Vw + lane * 16);
  v8f acc = {};
  for (int ch = 0; ch < 56; ++ch) {
    int s0c = ch * 32;
    int src = (s0c < 512) ? 0 : (s0c < 1024) ? 1 : (s0c < 1536) ? 2 : 3;
    int ls  = s0c - coff[src];
    // stage V[ls+lane][wave*16 .. +16) -> Vw[lane][0..16)
    const unsigned short* gsrc =
        Vt[src] + (((size_t)(b * H_ + h)) * Lsrc[src] + ls + lane) * HD_ + wave * 16;
    unsigned long long ga = (unsigned long long)(size_t)gsrc;
    asm volatile("global_load_async_to_lds_b128 %0, %1, off"
                 :: "v"(vw_lds), "v"(ga) : "memory");
    asm volatile("global_load_async_to_lds_b128 %0, %1, off offset:16"
                 :: "v"(vw_lds), "v"(ga) : "memory");
    asm volatile("s_wait_asynccnt 0" ::: "memory");

    Frag P, Vf;
    const unsigned short* prow = &Pb[lr * TCOLS + s0c];
    P.q2[0] = *(const uint4*)(prow + 8 * lh);               // contiguous bf16 cols
    P.q2[1] = *(const uint4*)(prow + 16 + 8 * lh);
#pragma unroll
    for (int e = 0; e < 8; ++e) {
      Vf.u[e]     = Vw[(8 * lh + e) * 16 + lr];             // B: d fixed, s varies
      Vf.u[8 + e] = Vw[(16 + 8 * lh + e) * 16 + lr];
    }
    acc = __builtin_amdgcn_wmma_f32_16x16x32_bf16(
        false, P.v, false, Vf.v, (short)0, acc, false, false);
  }
#pragma unroll
  for (int r = 0; r < 8; ++r)
    out[((size_t)(b * T_) + q0 + r + 8 * lh) * C_ + h * HD_ + wave * 16 + lr] = acc[r];
}

// ---------------------------------------------------------------------------
// Gated residual: out = x + mod[gate] * y
// ---------------------------------------------------------------------------
__global__ void residual_gate(const float* __restrict__ x,
                              const float* __restrict__ y,
                              const float* __restrict__ mod, int gate_idx,
                              float* __restrict__ out) {
  size_t idx = (size_t)blockIdx.x * 256 + threadIdx.x;   // B*T*C elements
  int bb = (int)(idx >> 19);                             // / (T_*C_)
  int c  = (int)(idx & (C_ - 1));
  out[idx] = x[idx] + mod[bb * (6 * C_) + gate_idx * C_ + c] * y[idx];
}

// ---------------------------------------------------------------------------
extern "C" void kernel_launch(void* const* d_in, const int* in_sizes, int n_in,
                              void* d_out, int out_size, void* d_ws, size_t ws_size,
                              hipStream_t stream) {
  (void)in_sizes; (void)n_in; (void)out_size; (void)ws_size;
  const float* x    = (const float*)d_in[0];
  const float* bev  = (const float*)d_in[1];
  const float* vlt  = (const float*)d_in[2];
  const float* rea  = (const float*)d_in[3];
  const float* cond = (const float*)d_in[4];
  const float* qpw = (const float*)d_in[5];  const float* qpb = (const float*)d_in[6];
  const float* abw = (const float*)d_in[7];  const float* abb = (const float*)d_in[8];
  const float* avw = (const float*)d_in[9];  const float* avb = (const float*)d_in[10];
  const float* arw = (const float*)d_in[11]; const float* arb = (const float*)d_in[12];
  const float* aow = (const float*)d_in[13]; const float* aob = (const float*)d_in[14];
  const float* ksw = (const float*)d_in[15]; const float* ksbb = (const float*)d_in[16];
  const float* vsw = (const float*)d_in[17]; const float* vsbb = (const float*)d_in[18];
  const float* kbw = (const float*)d_in[19]; const float* kbbb = (const float*)d_in[20];
  const float* vbw = (const float*)d_in[21]; const float* vbbb = (const float*)d_in[22];
  const float* kvw = (const float*)d_in[23]; const float* kvbb = (const float*)d_in[24];
  const float* vvw = (const float*)d_in[25]; const float* vvbb = (const float*)d_in[26];
  const float* krw = (const float*)d_in[27]; const float* krbb = (const float*)d_in[28];
  const float* vrw = (const float*)d_in[29]; const float* vrbb = (const float*)d_in[30];
  const float* ow  = (const float*)d_in[31]; const float* ob  = (const float*)d_in[32];
  const float* qnw = (const float*)d_in[33]; const float* knw = (const float*)d_in[34];
  const float* gating = (const float*)d_in[35];
  const float* tSelf = (const float*)d_in[36]; const float* tBev = (const float*)d_in[37];
  const float* tVl   = (const float*)d_in[38]; const float* tRe  = (const float*)d_in[39];
  const float* bSelf = (const float*)d_in[40]; const float* bBev = (const float*)d_in[41];
  const float* bVl   = (const float*)d_in[42]; const float* bRe  = (const float*)d_in[43];
  const float* flnw = (const float*)d_in[44]; const float* flnb = (const float*)d_in[45];
  const float* f1w  = (const float*)d_in[46]; const float* f1b  = (const float*)d_in[47];
  const float* f2w  = (const float*)d_in[48]; const float* f2b  = (const float*)d_in[49];
  const float* npw  = (const float*)d_in[50]; const float* npb  = (const float*)d_in[51];
  const float* adw  = (const float*)d_in[52]; const float* adb  = (const float*)d_in[53];

  // ----- workspace bump allocator -----
  char* wp = (char*)d_ws;
  auto alloc = [&](size_t bytes) -> void* {
    void* p = (void*)wp;
    wp += (bytes + 255) & ~(size_t)255;
    return p;
  };
  const size_t BT = (size_t)B_ * T_;                 // 2048 rows
  float* mod    = (float*)alloc((size_t)B_ * 6 * C_ * 4);
  float* x_norm = (float*)alloc(BT * C_ * 4);
  float* q_base = (float*)alloc(BT * C_ * 4);        // later reused as o_proj out
  float* qvar   = (float*)alloc(BT * C_ * 4);        // reused for bev/vl/reason q
  float* hidden = (float*)alloc(BT * 256 * 4);
  float* flat   = (float*)alloc(BT * C_ * 4);        // reused for all k/v flats
  float* attn_o = (float*)alloc(BT * C_ * 4);
  float* x1     = (float*)alloc(BT * C_ * 4);
  float* h1     = (float*)alloc(BT * (size_t)DFF_ * 4);
  float* h2     = (float*)alloc(BT * C_ * 4);
  const size_t HB = (size_t)B_ * H_ * T_ * HD_;      // 2M halfs for L=512
  const size_t HR = (size_t)B_ * H_ * TR_ * HD_;     // 1M halfs for L=256
  unsigned short* qSb = (unsigned short*)alloc(HB * 2);
  unsigned short* qBb = (unsigned short*)alloc(HB * 2);
  unsigned short* qVb = (unsigned short*)alloc(HB * 2);
  unsigned short* qRb = (unsigned short*)alloc(HB * 2);
  unsigned short* kSb = (unsigned short*)alloc(HB * 2);
  unsigned short* vSb = (unsigned short*)alloc(HB * 2);
  unsigned short* kBb = (unsigned short*)alloc(HB * 2);
  unsigned short* vBb = (unsigned short*)alloc(HB * 2);
  unsigned short* kVb = (unsigned short*)alloc(HB * 2);
  unsigned short* vVb = (unsigned short*)alloc(HB * 2);
  unsigned short* kRb = (unsigned short*)alloc(HR * 2);
  unsigned short* vRb = (unsigned short*)alloc(HR * 2);

  // ----- 1) adaLN modulation -----
  adaln_kernel<<<B_ * 24, 256, 0, stream>>>(cond, adw, adb, mod);

  // ----- 2) pre-norm + modulate -----
  ln_mod_kernel<<<BT, 256, 0, stream>>>(x, npw, npb, mod, /*scale*/1, /*shift*/0, x_norm);

  // ----- 3) q_proj and q adapters -----
  dim3 g16(16, 32), g4(4, 32);
  gemm_wmma<0><<<g16, 128, 0, stream>>>(x_norm, qpw, qpb, nullptr, q_base, 2048, 1024, 1024);

  // bev adapter: hidden = tanh(x_norm @ Wab + b); qvar = hidden @ Wout + b + q_base
  gemm_wmma<1><<<g4, 128, 0, stream>>>(x_norm, abw, abb, nullptr, hidden, 2048, 256, 1024);
  gemm_wmma<0><<<g16, 128, 0, stream>>>(hidden, aow, aob, q_base, qvar, 2048, 1024, 256);
  qk_prep<1><<<4096, 256, 0, stream>>>(qvar, qnw, qBb, T_);
  // vl adapter
  gemm_wmma<1><<<g4, 128, 0, stream>>>(x_norm, avw, avb, nullptr, hidden, 2048, 256, 1024);
  gemm_wmma<0><<<g16, 128, 0, stream>>>(hidden, aow, aob, q_base, qvar, 2048, 1024, 256);
  qk_prep<1><<<4096, 256, 0, stream>>>(qvar, qnw, qVb, T_);
  // reason adapter
  gemm_wmma<1><<<g4, 128, 0, stream>>>(x_norm, arw, arb, nullptr, hidden, 2048, 256, 1024);
  gemm_wmma<0><<<g16, 128, 0, stream>>>(hidden, aow, aob, q_base, qvar, 2048, 1024, 256);
  qk_prep<1><<<4096, 256, 0, stream>>>(qvar, qnw, qRb, T_);
  // base q
  qk_prep<1><<<4096, 256, 0, stream>>>(q_base, qnw, qSb, T_);

  // ----- 4) K/V projections + RMS/RoPE/bf16 prep -----
  gemm_wmma<0><<<g16, 128, 0, stream>>>(x_norm, ksw, ksbb, nullptr, flat, 2048, 1024, 1024);
  qk_prep<1><<<4096, 256, 0, stream>>>(flat, knw, kSb, T_);
  gemm_wmma<0><<<g16, 128, 0, stream>>>(x_norm, vsw, vsbb, nullptr, flat, 2048, 1024, 1024);
  qk_prep<0><<<4096, 256, 0, stream>>>(flat, knw, vSb, T_);
  gemm_wmma<0><<<g16, 128, 0, stream>>>(bev, kbw, kbbb, nullptr, flat, 2048, 1024, 1024);
  qk_prep<1><<<4096, 256, 0, stream>>>(flat, knw, kBb, T_);
  gemm_wmma<0><<<g16, 128, 0, stream>>>(bev, vbw, vbbb, nullptr, flat, 2048, 1024, 1024);
  qk_prep<0><<<4096, 256, 0, stream>>>(flat, knw, vBb, T_);
  gemm_wmma<0><<<g16, 128, 0, stream>>>(vlt, kvw, kvbb, nullptr, flat, 2048, 1024, 1024);
  qk_prep<1><<<4096, 256, 0, stream>>>(flat, knw, kVb, T_);
  gemm_wmma<0><<<g16, 128, 0, stream>>>(vlt, vvw, vvbb, nullptr, flat, 2048, 1024, 1024);
  qk_prep<0><<<4096, 256, 0, stream>>>(flat, knw, vVb, T_);
  dim3 g16r(16, 16);
  gemm_wmma<0><<<g16r, 128, 0, stream>>>(rea, krw, krbb, nullptr, flat, 1024, 1024, 1024);
  qk_prep<1><<<2048, 256, 0, stream>>>(flat, knw, kRb, TR_);
  gemm_wmma<0><<<g16r, 128, 0, stream>>>(rea, vrw, vrbb, nullptr, flat, 1024, 1024, 1024);
  qk_prep<0><<<2048, 256, 0, stream>>>(flat, knw, vRb, TR_);

  // ----- 5) fused multi-source attention -----
  const int attn_lds = 16 * SROW * 4 + 16 * TCOLS * 2 + 4 * 32 * 16 * 2;
  attn_kernel<<<dim3(T_ / 16, H_, B_), 128, attn_lds, stream>>>(
      qSb, qBb, qVb, qRb, kSb, kBb, kVb, kRb, vSb, vBb, vVb, vRb,
      gating, tSelf, tBev, tVl, tRe, bSelf, bBev, bVl, bRe, attn_o);

  // ----- 6) o_proj + gated residual -----
  gemm_wmma<0><<<g16, 128, 0, stream>>>(attn_o, ow, ob, nullptr, q_base, 2048, 1024, 1024);
  residual_gate<<<8192, 256, 0, stream>>>(x, q_base, mod, /*gate_attn*/2, x1);

  // ----- 7) FFN -----
  ln_mod_kernel<<<BT, 256, 0, stream>>>(x1, flnw, flnb, mod, /*scale*/4, /*shift*/3, x_norm);
  gemm_wmma<2><<<dim3(64, 32), 128, 0, stream>>>(x_norm, f1w, f1b, nullptr, h1, 2048, 4096, 1024);
  gemm_wmma<0><<<g16, 128, 0, stream>>>(h1, f2w, f2b, nullptr, h2, 2048, 1024, 4096);
  residual_gate<<<8192, 256, 0, stream>>>(x1, h2, mod, /*gate_ffn*/5, (float*)d_out);
}
